// fptc_gnn_33655363732143
// MI455X (gfx1250) — matrix operand order repaired and checked by reading, not compiled, not run on hardware
//
#include <hip/hip_runtime.h>

#define NPER_ 131072
#define H_ 128
#define FEAT_ 32
#define NLEV_ 8

typedef unsigned int u32;
typedef unsigned short u16;
typedef __attribute__((ext_vector_type(16))) __bf16 v16bf;
typedef __attribute__((ext_vector_type(8)))  float  v8f;
typedef __attribute__((ext_vector_type(4)))  u32    u32x4;

union Frag { v16bf bf; u32x4 q[2]; };

__device__ __forceinline__ u16 f2bf(float f) {
  u32 x = __builtin_bit_cast(u32, f);
  x += 0x7fffu + ((x >> 16) & 1u);     // round-to-nearest-even
  return (u16)(x >> 16);
}
__device__ __forceinline__ float bf2f(u16 h) {
  u32 x = (u32)h << 16;
  return __builtin_bit_cast(float, x);
}

// Branch-free transcendentals on the hardware TRANS pipe (co-issues with WMMA).
__device__ __forceinline__ float fast_tanh(float x) {
  // tanh(x) = 1 - 2/(exp(2x)+1); exp2 saturates -> correct +-1 tails.
  float e = __builtin_amdgcn_exp2f(x * 2.8853900817779268f);  // 2*log2(e)
  return 1.f - 2.f * __builtin_amdgcn_rcpf(e + 1.f);
}
__device__ __forceinline__ float fast_sigmoid(float x) {
  float e = __builtin_amdgcn_exp2f(-1.4426950408889634f * x);
  return __builtin_amdgcn_rcpf(1.f + e);
}
__device__ __forceinline__ float fast_exp(float x) {
  return __builtin_amdgcn_exp2f(1.4426950408889634f * x);
}

// A fragment (16x32 bf16): lane row = lane&15; hi = lane>>4 selects K-halves.
__device__ __forceinline__ v16bf load_a(const u16* row, int kc, int hi) {
  const u16* p = row + kc + hi * 8;
  Frag f;
  f.q[0] = *(const u32x4*)(p);
  f.q[1] = *(const u32x4*)(p + 16);
  return f.bf;
}
// B fragment (32x16 bf16) from W^T row-major [N][K]: lane col = lane&15,
// hi selects K-halves: 8 halves at kc + hi*8 and 8 at kc + 16 + hi*8.
__device__ __forceinline__ v16bf load_b(const u16* wrow, int kc, int hi) {
  const u16* p = wrow + kc + hi * 8;
  Frag f;
  f.q[0] = *(const u32x4*)(p);
  f.q[1] = *(const u32x4*)(p + 16);
  return f.bf;
}

#define WMMA_BF16(a, b, c) \
  __builtin_amdgcn_wmma_f32_16x16x32_bf16(false, (a), false, (b), (short)0, (c), false, false)

// D layout: VGPR j -> row j + 8*hi, lane&15 -> col. Apply bias+activation, store bf16 to tile.
__device__ __forceinline__ void act_store(u16* tile, int col, int hi, v8f acc, float bv, bool relu_) {
#pragma unroll
  for (int j = 0; j < 8; ++j) {
    float v = acc[j] + bv;
    v = relu_ ? fmaxf(v, 0.f) : fast_tanh(v);
    tile[(j + hi * 8) * H_ + col] = f2bf(v);
  }
}

// One wave computes a 16x128 output tile: preload KCH A-fragments, sweep 8 N-tiles.
template <int KCH, class AF, class DW>
__device__ __forceinline__ void gemm_stage(AF&& loadA, const u16* WT, DW&& writeD, int lane) {
  const int hi = lane >> 4, ln = lane & 15;
  v16bf afr[KCH];
#pragma unroll
  for (int kc = 0; kc < KCH; ++kc) afr[kc] = loadA(kc, hi);
#pragma unroll
  for (int nt = 0; nt < 8; ++nt) {
    const int col = nt * 16 + ln;
    const u16* wrow = WT + (size_t)col * (KCH * 32);
    v8f acc = {};
#pragma unroll
    for (int kc = 0; kc < KCH; ++kc)
      acc = WMMA_BF16(afr[kc], load_b(wrow, kc * 32, hi), acc);
    writeD(col, hi, acc);
  }
}

__device__ __forceinline__ void project_row(const u16* rowp, const float* Wp, const float* bp,
                                            float* outRow) {
  float s0 = bp[0], s1 = bp[1], s2 = bp[2];
#pragma unroll 4
  for (int k = 0; k < H_; ++k) {
    float x = bf2f(rowp[k]);
    s0 += x * Wp[k * 3 + 0];
    s1 += x * Wp[k * 3 + 1];
    s2 += x * Wp[k * 3 + 2];
  }
  s0 = fast_sigmoid(s0);
  s1 = fast_sigmoid(s1);
  s2 = fast_sigmoid(s2);
  float m = fmaxf(s0, fmaxf(s1, s2));
  float e0 = fast_exp(s0 - m), e1 = fast_exp(s1 - m), e2 = fast_exp(s2 - m);
  float inv = __builtin_amdgcn_rcpf(e0 + e1 + e2);
  outRow[0] = e0 * inv; outRow[1] = e1 * inv; outRow[2] = e2 * inv;
}

// ---------------- prep kernels ----------------
__global__ void fptc_cvt_bf16(const float* __restrict__ src, u16* __restrict__ dst, int n) {
  int i = blockIdx.x * blockDim.x + threadIdx.x;
  if (i < n) dst[i] = f2bf(src[i]);
}
__global__ void fptc_transpose_bf16(const float* __restrict__ src, u16* __restrict__ dst,
                                    int K, int N) {
  int i = blockIdx.x * blockDim.x + threadIdx.x;
  if (i < K * N) {
    int k = i / N, n = i - k * N;
    dst[n * K + k] = f2bf(src[i]);
  }
}

// ---------------- level 0 ----------------
__global__ __launch_bounds__(128) void fptc_level0_kernel(
    const u16* __restrict__ featsB0, u16* __restrict__ cur, const u16* __restrict__ WopT,
    const float* __restrict__ b_op, const float* __restrict__ Wp, const float* __restrict__ bp,
    float* __restrict__ outL) {
  __shared__ __align__(16) u16 ldsT[4][16 * H_];
  const int lane = threadIdx.x & 31;
  const int wv = threadIdx.x >> 5;
  const int ln = lane & 15;
  const int rowBase = blockIdx.x * 64 + wv * 16;
  u16* T0 = ldsT[wv];

  const u16* frow = featsB0 + (size_t)(rowBase + ln) * FEAT_;
  gemm_stage<1>([&](int, int h) -> v16bf { return load_a(frow, 0, h); }, WopT,
                [&](int col, int h, v8f acc) { act_store(T0, col, h, acc, b_op[col], false); },
                lane);
  __builtin_amdgcn_wave_barrier();

  {
    const u32x4* src = (const u32x4*)T0;
    u32x4* dst = (u32x4*)(cur + (size_t)rowBase * H_);
#pragma unroll
    for (int i = 0; i < 8; ++i) dst[lane + i * 32] = src[lane + i * 32];
  }
  if (lane < 16)
    project_row(T0 + lane * H_, Wp, bp, outL + (size_t)(rowBase + lane) * 3);
}

// ---------------- levels 1..7 ----------------
__global__ __launch_bounds__(128) void fptc_level_kernel(
    const u16* __restrict__ featsBt, const u16* __restrict__ prev, u16* __restrict__ cur,
    const int* __restrict__ parents2, const int* __restrict__ isun, const u16* __restrict__ WT,
    const float* __restrict__ b_op, const float* __restrict__ bn0, const float* __restrict__ bn1,
    const float* __restrict__ bu0, const float* __restrict__ bu1, const float* __restrict__ bb0,
    const float* __restrict__ Wp, const float* __restrict__ bp, float* __restrict__ outL) {
  __shared__ __align__(16) u16 ldsT[3][4][16 * H_];
  const int lane = threadIdx.x & 31;
  const int wv = threadIdx.x >> 5;
  const int hi = lane >> 4, ln = lane & 15;
  const int rowBase = blockIdx.x * 64 + wv * 16;

  u16* T0 = ldsT[0][wv];
  u16* T1 = ldsT[1][wv];
  u16* T2 = ldsT[2][wv];

  const u16* WopT = WT;              // [128][32]
  const u16* Wn0T = WT + 4096;       // [128][256]
  const u16* Wn1T = WT + 36864;      // [128][128]
  const u16* Wu0T = WT + 53248;      // [128][128]
  const u16* Wu1T = WT + 69632;      // [128][128]
  const u16* Wb0T = WT + 86016;      // [128][256]

  // gathered parent rows (bf16 rows of previous level, L2-resident)
  const int gr = rowBase + ln;
  const int p0 = parents2[gr * 2 + 0];
  const int p1 = parents2[gr * 2 + 1];
  const u16* m0row = prev + (size_t)p0 * H_;
  const u16* m1row = prev + (size_t)p1 * H_;

  // hu = tanh(m0 @ Wu0 + bu0) -> T0
  gemm_stage<4>([&](int kc, int h) -> v16bf { return load_a(m0row, kc * 32, h); }, Wu0T,
                [&](int col, int h, v8f acc) { act_store(T0, col, h, acc, bu0[col], false); },
                lane);
  // hb = tanh([m0,m1] @ Wb0 + bb0) -> T1   (K=256)
  gemm_stage<8>([&](int kc, int h) -> v16bf {
                  return (kc < 4) ? load_a(m0row, kc * 32, h) : load_a(m1row, (kc - 4) * 32, h);
                }, Wb0T,
                [&](int col, int h, v8f acc) { act_store(T1, col, h, acc, bb0[col], false); },
                lane);
  __builtin_amdgcn_wave_barrier();

  // redux = sel(relu(T0@Wu1+bu1), tanh(T1@Wu1+bu1)) -> T2  (shared B fragments)
  {
    v16bf a0[4], a1[4];
    const u16* t0r = T0 + ln * H_;
    const u16* t1r = T1 + ln * H_;
#pragma unroll
    for (int kc = 0; kc < 4; ++kc) {
      a0[kc] = load_a(t0r, kc * 32, hi);
      a1[kc] = load_a(t1r, kc * 32, hi);
    }
    int unm = 0;
#pragma unroll
    for (int j = 0; j < 8; ++j)
      unm |= (isun[rowBase + j + hi * 8] == 1) ? (1 << j) : 0;
#pragma unroll
    for (int nt = 0; nt < 8; ++nt) {
      const int col = nt * 16 + ln;
      const u16* wrow = Wu1T + (size_t)col * H_;
      v8f c1 = {}, c2 = {};
#pragma unroll
      for (int kc = 0; kc < 4; ++kc) {
        v16bf b = load_b(wrow, kc * 32, hi);
        c1 = WMMA_BF16(a0[kc], b, c1);
        c2 = WMMA_BF16(a1[kc], b, c2);
      }
      const float bv = bu1[col];
#pragma unroll
      for (int j = 0; j < 8; ++j) {
        float hu2 = fmaxf(c1[j] + bv, 0.f);
        float hb2 = fast_tanh(c2[j] + bv);
        T2[(j + hi * 8) * H_ + col] = f2bf(((unm >> j) & 1) ? hu2 : hb2);
      }
    }
  }
  __builtin_amdgcn_wave_barrier();

  // ne = tanh(feats @ Wop + b_op) -> T0
  const u16* frow = featsBt + (size_t)gr * FEAT_;
  gemm_stage<1>([&](int, int h) -> v16bf { return load_a(frow, 0, h); }, WopT,
                [&](int col, int h, v8f acc) { act_store(T0, col, h, acc, b_op[col], false); },
                lane);
  __builtin_amdgcn_wave_barrier();

  // ne2 = tanh([T0,T2] @ Wn0 + bn0) -> T1   (K=256)
  {
    const u16* t0r = T0 + ln * H_;
    const u16* t2r = T2 + ln * H_;
    gemm_stage<8>([&](int kc, int h) -> v16bf {
                    return (kc < 4) ? load_a(t0r, kc * 32, h) : load_a(t2r, (kc - 4) * 32, h);
                  }, Wn0T,
                  [&](int col, int h, v8f acc) { act_store(T1, col, h, acc, bn0[col], false); },
                  lane);
  }
  __builtin_amdgcn_wave_barrier();

  // ne3 = tanh(T1 @ Wn1 + bn1) -> T0
  {
    const u16* t1r = T1 + ln * H_;
    gemm_stage<4>([&](int kc, int h) -> v16bf { return load_a(t1r, kc * 32, h); }, Wn1T,
                  [&](int col, int h, v8f acc) { act_store(T0, col, h, acc, bn1[col], false); },
                  lane);
  }
  __builtin_amdgcn_wave_barrier();

  // write bf16 activations for next level (contiguous b128 stores)
  {
    const u32x4* src = (const u32x4*)T0;
    u32x4* dst = (u32x4*)(cur + (size_t)rowBase * H_);
#pragma unroll
    for (int i = 0; i < 8; ++i) dst[lane + i * 32] = src[lane + i * 32];
  }
  // projection + sigmoid + softmax
  if (lane < 16)
    project_row(T0 + lane * H_, Wp, bp, outL + (size_t)(rowBase + lane) * 3);
}

extern "C" void kernel_launch(void* const* d_in, const int* in_sizes, int n_in,
                              void* d_out, int out_size, void* d_ws, size_t ws_size,
                              hipStream_t stream) {
  (void)in_sizes; (void)n_in; (void)out_size; (void)ws_size;
  const float* node_feats = (const float*)d_in[0];
  const int*   parents    = (const int*)d_in[1];
  const int*   is_unary   = (const int*)d_in[2];
  const float* W_op = (const float*)d_in[3];
  const float* b_op = (const float*)d_in[4];
  const float* Wn0  = (const float*)d_in[5];
  const float* bn0  = (const float*)d_in[6];
  const float* Wn1  = (const float*)d_in[7];
  const float* bn1  = (const float*)d_in[8];
  const float* Wu0  = (const float*)d_in[9];
  const float* bu0  = (const float*)d_in[10];
  const float* Wu1  = (const float*)d_in[11];
  const float* bu1  = (const float*)d_in[12];
  const float* Wb0  = (const float*)d_in[13];
  const float* bb0  = (const float*)d_in[14];
  const float* Wp   = (const float*)d_in[15];
  const float* bp   = (const float*)d_in[16];
  float* out = (float*)d_out;

  // workspace layout (bf16): feats | bufA | bufB | packed W^T
  u16* featsB = (u16*)d_ws;
  u16* bufA = featsB + (size_t)NLEV_ * NPER_ * FEAT_;
  u16* bufB = bufA + (size_t)NPER_ * H_;
  u16* WT   = bufB + (size_t)NPER_ * H_;

  {
    int n = NLEV_ * NPER_ * FEAT_;
    fptc_cvt_bf16<<<(n + 255) / 256, 256, 0, stream>>>(node_feats, featsB, n);
    fptc_transpose_bf16<<<(4096  + 255) / 256, 256, 0, stream>>>(W_op, WT + 0,     FEAT_,  H_);
    fptc_transpose_bf16<<<(32768 + 255) / 256, 256, 0, stream>>>(Wn0,  WT + 4096,  2 * H_, H_);
    fptc_transpose_bf16<<<(16384 + 255) / 256, 256, 0, stream>>>(Wn1,  WT + 36864, H_,     H_);
    fptc_transpose_bf16<<<(16384 + 255) / 256, 256, 0, stream>>>(Wu0,  WT + 53248, H_,     H_);
    fptc_transpose_bf16<<<(16384 + 255) / 256, 256, 0, stream>>>(Wu1,  WT + 69632, H_,     H_);
    fptc_transpose_bf16<<<(32768 + 255) / 256, 256, 0, stream>>>(Wb0,  WT + 86016, 2 * H_, H_);
  }

  fptc_level0_kernel<<<NPER_ / 64, 128, 0, stream>>>(featsB, bufA, WT, b_op, Wp, bp, out);

  u16* prev = bufA;
  u16* cur  = bufB;
  for (int t = 1; t < NLEV_; ++t) {
    fptc_level_kernel<<<NPER_ / 64, 128, 0, stream>>>(
        featsB + (size_t)t * NPER_ * FEAT_, prev, cur,
        parents + (size_t)(t - 1) * NPER_ * 2, is_unary + (size_t)(t - 1) * NPER_,
        WT, b_op, bn0, bn1, bu0, bu1, bb0, Wp, bp,
        out + (size_t)t * NPER_ * 3);
    u16* tmp = prev; prev = cur; cur = tmp;
  }
}
